// SimpleGNNClassifier_10677288698291
// MI455X (gfx1250) — matrix operand order repaired and checked by reading, not compiled, not run on hardware
//
#include <hip/hip_runtime.h>

typedef __attribute__((ext_vector_type(16))) _Float16 v16h;
typedef __attribute__((ext_vector_type(8)))  _Float16 v8h;
typedef __attribute__((ext_vector_type(8)))  float    v8f;

#define NNODES 100000
#define NEDGES 1600000
#define INDIM  165
#define KPAD1  192      // 165 padded to multiple of 32
#define HID    128
#define NBATCH 50000

// ---------------- utility kernels ----------------

__global__ void k_fill(float* __restrict__ p, float v, long long n) {
  long long i = (long long)blockIdx.x * blockDim.x + threadIdx.x;
  if (i < n) p[i] = v;
}

__global__ void k_zero1(float* p) {
  if (blockIdx.x == 0 && threadIdx.x == 0) p[0] = 0.0f;
}

// deg[dst] += 1 over edges (deg pre-initialized to 1.0 for self-loops)
__global__ void k_deg(const int* __restrict__ dst, float* __restrict__ deg) {
  int e = blockIdx.x * blockDim.x + threadIdx.x;
  if (e < NEDGES) unsafeAtomicAdd(&deg[dst[e]], 1.0f);
}

__global__ void k_rsqrt_inplace(float* __restrict__ d) {
  int i = blockIdx.x * blockDim.x + threadIdx.x;
  if (i < NNODES) d[i] = rsqrtf(d[i]);   // deg >= 1 always
}

// ---------------- preprocessing: pack to f16 ----------------

// Pack W[K,N] (f32, row-major) into WMMA B-fragment order, K zero-padded to KP:
// out[((kb*(N/16) + nt)*32 + lane)*16 + j] = W[kb*32 + (lane>>4)*8 + (j&7) + (j>>3)*16][nt*16 + (lane&15)]
__global__ void k_pack_w(const float* __restrict__ W, _Float16* __restrict__ out,
                         int K, int KP, int N) {
  const int tid = blockIdx.x * blockDim.x + threadIdx.x;
  const int ntN = N >> 4;
  const int total = (KP >> 5) * ntN * 32 * 16;
  if (tid >= total) return;
  const int j    = tid & 15;
  const int lane = (tid >> 4) & 31;
  const int t2   = tid >> 9;            // kb*ntN + nt
  const int kb   = t2 / ntN;
  const int nt   = t2 - kb * ntN;
  const int n    = nt * 16 + (lane & 15);
  const int k    = kb * 32 + (lane >> 4) * 8 + (j & 7) + (j >> 3) * 16;
  const float v  = (k < K) ? W[(size_t)k * N + n] : 0.0f;
  out[tid] = (_Float16)v;
}

// x [N, INDIM] f32 -> xh [N, KPAD1] f16 (zero-padded rows)
__global__ void k_pack_x(const float* __restrict__ x, _Float16* __restrict__ xh) {
  long long i = (long long)blockIdx.x * blockDim.x + threadIdx.x;
  if (i >= (long long)NNODES * KPAD1) return;
  const int  j = (int)(i % KPAD1);
  const long long r = i / KPAD1;
  xh[i] = (j < INDIM) ? (_Float16)x[r * INDIM + j] : (_Float16)0.0f;
}

// ---------------- WMMA GEMM: C[M,N] = Ah[M,KP] @ Wp (+bias, relu) ----------------
// f16 inputs (pre-packed), f32 accumulate/output. One wave per 16x16 tile,
// 4 waves/block. KP multiple of 32, M multiple of 16. No branches in K-loop.
__global__ __launch_bounds__(128)
void k_gemm_wmma(const _Float16* __restrict__ Ah, const _Float16* __restrict__ Wp,
                 float* __restrict__ C, const float* __restrict__ bias,
                 int M, int KP, int N, int relu) {
  const int lane = threadIdx.x & 31;
  const int ntN  = N >> 4;
  const int tile = blockIdx.x * 4 + (threadIdx.x >> 5);
  if (tile >= (M >> 4) * ntN) return;   // wave-uniform exit: EXEC all-ones at WMMA
  const int mt = tile / ntN;
  const int nt = tile - mt * ntN;
  const int mbase = mt << 4, nbase = nt << 4;
  const int hi = lane >> 4;             // half-wave K sub-block select
  const int lo = lane & 15;

  const _Float16* __restrict__ arow = Ah + (size_t)(mbase + lo) * (size_t)KP;
  const _Float16* __restrict__ wp   = Wp + ((size_t)nt * 32 + lane) * 16;
  const size_t wstride = (size_t)ntN * 32 * 16;   // halves per k-tile of packed W

  v8f acc = {};
  const int kTiles = KP >> 5;
  for (int kb = 0; kb < kTiles; ++kb) {
    const int k0 = kb << 5;
    // A fragment (16x32 f16): two contiguous 16B groups per lane
    union { v16h v; struct { v8h l, h; } p; } af;
    af.p.l = *(const v8h*)(arow + k0 + hi * 8);        // halves 0..7
    af.p.h = *(const v8h*)(arow + k0 + 16 + hi * 8);   // halves 8..15
    // B fragment: one contiguous 32B per lane from packed weights
    const v16h bf = *(const v16h*)(wp + (size_t)kb * wstride);
    acc = __builtin_amdgcn_wmma_f32_16x16x32_f16(false, af.v, false, bf,
                                                 (short)0, acc, false, false);
  }

  // C/D layout: element r -> row mbase + r + 8*hi, col nbase + lo
  const int col = nbase + lo;
  const float bv = bias ? bias[col] : 0.0f;
#pragma unroll
  for (int r = 0; r < 8; ++r) {
    float v = acc[r] + bv;
    if (relu) v = fmaxf(v, 0.0f);
    C[(size_t)(mbase + r + hi * 8) * (size_t)N + col] = v;
  }
}

// ---------------- normalized gather-scatter aggregation ----------------
// segments: [0,E) edges, [E, E+N) self-loops. One wave per segment, 4 feats/lane.
__global__ void k_scatter(const float* __restrict__ H, const int* __restrict__ src,
                          const int* __restrict__ dst, const float* __restrict__ dinv,
                          float* __restrict__ agg) {
  long long t   = (long long)blockIdx.x * blockDim.x + threadIdx.x;
  long long seg = t >> 5;
  const int f   = (int)(t & 31);
  const long long nseg = (long long)NEDGES + NNODES;
  if (seg >= nseg) return;
  int s, d;
  if (seg < NEDGES) { s = src[seg]; d = dst[seg]; }
  else              { s = d = (int)(seg - NEDGES); }
  const float w = dinv[s] * dinv[d];
  const float* hs = H   + (size_t)s * HID;
  float*       ad = agg + (size_t)d * HID;
#pragma unroll
  for (int q = 0; q < 4; ++q) {
    const int c = f + 32 * q;
    unsafeAtomicAdd(&ad[c], hs[c] * w);
  }
}

// agg f32 + bias, ReLU, emit f16 activations for the next WMMA GEMM
__global__ void k_bias_relu_h(const float* __restrict__ agg,
                              const float* __restrict__ bias,
                              _Float16* __restrict__ out, long long n) {
  long long i = (long long)blockIdx.x * blockDim.x + threadIdx.x;
  if (i < n)
    out[i] = (_Float16)fmaxf(agg[i] + bias[(int)(i & (HID - 1))], 0.0f);
}

// ---------------- head: d @ Wd2 + bd2, softmax -> preds[N,2] ----------------
__global__ void k_head(const float* __restrict__ D, const float* __restrict__ Wd2,
                       const float* __restrict__ bd2, float* __restrict__ preds) {
  int i = blockIdx.x * blockDim.x + threadIdx.x;
  if (i >= NNODES) return;
  const float* row = D + (size_t)i * HID;
  float l0 = bd2[0], l1 = bd2[1];
#pragma unroll 4
  for (int k = 0; k < HID; ++k) {
    const float v = row[k];
    l0 += v * Wd2[2 * k];
    l1 += v * Wd2[2 * k + 1];
  }
  const float m  = fmaxf(l0, l1);
  const float e0 = __expf(l0 - m), e1 = __expf(l1 - m);
  const float inv = 1.0f / (e0 + e1);
  preds[2 * i]     = e0 * inv;
  preds[2 * i + 1] = e1 * inv;
}

// CE over already-softmaxed preds (faithful double-softmax), mean over batch
__global__ void k_loss(const float* __restrict__ preds, const int* __restrict__ batch,
                       const int* __restrict__ labels, float* __restrict__ out) {
  int j = blockIdx.x * blockDim.x + threadIdx.x;
  if (j >= NBATCH) return;
  const int node = batch[j];
  const int t    = labels[node];
  const float p0 = preds[2 * node], p1 = preds[2 * node + 1];
  const float m   = fmaxf(p0, p1);
  const float lse = m + __logf(__expf(p0 - m) + __expf(p1 - m));
  const float lp  = (t ? p1 : p0) - lse;
  unsafeAtomicAdd(out, -lp * (1.0f / NBATCH));
}

// ---------------- launcher ----------------
extern "C" void kernel_launch(void* const* d_in, const int* in_sizes, int n_in,
                              void* d_out, int out_size, void* d_ws, size_t ws_size,
                              hipStream_t stream) {
  const float* x    = (const float*)d_in[0];
  const float* W1   = (const float*)d_in[1];
  const float* b1   = (const float*)d_in[2];
  const float* W2   = (const float*)d_in[3];
  const float* b2   = (const float*)d_in[4];
  const float* Wd1  = (const float*)d_in[5];
  const float* bd1  = (const float*)d_in[6];
  const float* Wd2  = (const float*)d_in[7];
  const float* bd2  = (const float*)d_in[8];
  const int*   ei   = (const int*)d_in[9];
  const int*   bat  = (const int*)d_in[10];
  const int*   lab  = (const int*)d_in[11];
  const int*   src  = ei;
  const int*   dst  = ei + NEDGES;

  float* ws = (float*)d_ws;
  const size_t NH = (size_t)NNODES * HID;
  float*     A     = ws;                         // [N,HID] f32  (GEMM out / scatter in)
  float*     B     = A + NH;                     // [N,HID] f32  (aggregation)
  _Float16*  Xh    = (_Float16*)(B + NH);        // [N,KPAD1] f16; activations alias it
  _Float16*  Hh    = Xh;                         // [N,HID]   f16 (reuses dead Xh space)
  float*     after = (float*)(Xh + (size_t)NNODES * KPAD1);
  _Float16*  W1p   = (_Float16*)after;           // 192*128 halves
  _Float16*  W2p   = W1p + (size_t)KPAD1 * HID;  // 128*128 halves
  _Float16*  Wd1p  = W2p + (size_t)HID * HID;    // 128*128 halves
  float*     dinv  = (float*)(Wd1p + (size_t)HID * HID);
  float*     preds = dinv + NNODES;              // [N,2]
  float*     loss  = (float*)d_out;

  const int T = 256;
  const int nhBlocks   = (int)((NH + T - 1) / T);
  const int nodeBlocks = (NNODES + T - 1) / T;
  const int edgeBlocks = (NEDGES + T - 1) / T;
  const long long segThreads = ((long long)NEDGES + NNODES) * 32;
  const int segBlocks  = (int)((segThreads + T - 1) / T);
  const int gemmTiles  = (NNODES / 16) * (HID / 16);
  const int gemmBlocks = (gemmTiles + 3) / 4;
  const long long xhN  = (long long)NNODES * KPAD1;

  // pack weights + features to f16
  k_pack_w<<<((KPAD1 / 32) * (HID / 16) * 512 + T - 1) / T, T, 0, stream>>>(W1, W1p, INDIM, KPAD1, HID);
  k_pack_w<<<((HID / 32) * (HID / 16) * 512 + T - 1) / T, T, 0, stream>>>(W2, W2p, HID, HID, HID);
  k_pack_w<<<((HID / 32) * (HID / 16) * 512 + T - 1) / T, T, 0, stream>>>(Wd1, Wd1p, HID, HID, HID);
  k_pack_x<<<(int)((xhN + T - 1) / T), T, 0, stream>>>(x, Xh);

  // degree -> dinv
  k_fill<<<nodeBlocks, T, 0, stream>>>(dinv, 1.0f, NNODES);   // self-loop
  k_deg<<<edgeBlocks, T, 0, stream>>>(dst, dinv);
  k_rsqrt_inplace<<<nodeBlocks, T, 0, stream>>>(dinv);

  // layer 1: A = x @ W1 ; B = scatter(A) ; Hh = relu(B + b1)   (Hh reuses Xh)
  k_gemm_wmma<<<gemmBlocks, 128, 0, stream>>>(Xh, W1p, A, nullptr, NNODES, KPAD1, HID, 0);
  k_fill<<<nhBlocks, T, 0, stream>>>(B, 0.0f, (long long)NH);
  k_scatter<<<segBlocks, T, 0, stream>>>(A, src, dst, dinv, B);
  k_bias_relu_h<<<nhBlocks, T, 0, stream>>>(B, b1, Hh, (long long)NH);

  // layer 2: A = Hh @ W2 ; B = scatter(A) ; Hh = relu(B + b2)
  k_gemm_wmma<<<gemmBlocks, 128, 0, stream>>>(Hh, W2p, A, nullptr, NNODES, HID, HID, 0);
  k_fill<<<nhBlocks, T, 0, stream>>>(B, 0.0f, (long long)NH);
  k_scatter<<<segBlocks, T, 0, stream>>>(A, src, dst, dinv, B);
  k_bias_relu_h<<<nhBlocks, T, 0, stream>>>(B, b2, Hh, (long long)NH);

  // head: A = relu(Hh @ Wd1 + bd1) (fused) ; preds = softmax(A @ Wd2 + bd2)
  k_gemm_wmma<<<gemmBlocks, 128, 0, stream>>>(Hh, Wd1p, A, bd1, NNODES, HID, HID, 1);
  k_head<<<nodeBlocks, T, 0, stream>>>(A, Wd2, bd2, preds);

  // loss
  k_zero1<<<1, 1, 0, stream>>>(loss);
  k_loss<<<(NBATCH + T - 1) / T, T, 0, stream>>>(preds, bat, lab, loss);
}
